// TriDimensionalAttentionBlock_22170621182173
// MI455X (gfx1250) — matrix operand order repaired and checked by reading, not compiled, not run on hardware
//
#include <hip/hip_runtime.h>
#include <hip/hip_bf16.h>
#include <math.h>

typedef __attribute__((ext_vector_type(16))) __bf16 v16bf;
typedef __attribute__((ext_vector_type(8)))  __bf16 v8bf;
typedef __attribute__((ext_vector_type(8)))  float  v8f;

#define K_H     128
#define K_DOUT  256
#define K_HEADS 8
#define K_KD    32
#define K_B     8
#define K_N     512
#define K_I     3
#define K_O     2
#define K_TOK   (K_B*K_N*K_I*K_O)   /* 24576 */
#define K_NCAT  (9*K_DOUT)          /* 2304  */
#define K_KOUT  (3*K_DOUT)          /* 768   */

#define INV_SQRT_KD 0.17677669529663687f
#define INV_SQRT2   0.7071067811865476f

// token(b,n,i,o) = b*3072 + n*6 + i*2 + o  (row-major over [B,N,I,O])

__device__ __forceinline__ v16bf combine8(v8bf lo, v8bf hi) {
  v16bf r;
#pragma unroll
  for (int i = 0; i < 8; ++i) { r[i] = lo[i]; r[i + 8] = hi[i]; }
  return r;
}

// A-fragment (16x32 bf16, M x K) from row-major [M][stride] memory.
// Layout: lanes 0-15: row=lane, K = 0..15; lanes 16-31: row=lane-16, K = 16..31 (matches B fragment).
__device__ __forceinline__ v16bf load_a_frag(const __bf16* base, int stride, int m0, int k0, int lane) {
  const int row  = m0 + (lane & 15);
  const int koff = (lane < 16) ? 0 : 16;
  const __bf16* p = base + (long)row * stride + k0 + koff;
  return combine8(*(const v8bf*)p, *(const v8bf*)(p + 8));
}

// B-fragment (32x16 bf16, K x N) from [N][stride] (i.e. transposed-weight) memory.
// Layout (contiguous k order, not ISA table order): lanes 0-15: col=lane, K=0..15; lanes 16-31: col=lane-16, K=16..31 -> one contiguous 32B run.
__device__ __forceinline__ v16bf load_b_frag(const __bf16* base, int stride, int n0, int k0, int lane) {
  const int col  = n0 + (lane & 15);
  const int koff = (lane < 16) ? 0 : 16;
  return *(const v16bf*)(base + (long)col * stride + k0 + koff);
}

__device__ __forceinline__ float gelu_tanh(float x) {
  float u = 0.7978845608028654f * (x + 0.044715f * x * x * x);
  return 0.5f * x * (1.0f + tanhf(u));
}

// ---------------- prep kernels ----------------

__global__ void prep_temb_kernel(const float* __restrict__ t, const float* __restrict__ wt,
                                 const float* __restrict__ bt, float* __restrict__ temb) {
  int b = blockIdx.x, e = threadIdx.x;
  float acc = bt[e];
  for (int k = 0; k < K_H; ++k) acc += t[b * K_H + k] * wt[k * K_H + e];
  temb[b * K_H + e] = acc;
}

__global__ void prep_y_kernel(const float* __restrict__ s, const float* __restrict__ temb,
                              __bf16* __restrict__ Y) {
  int idx = blockIdx.x * blockDim.x + threadIdx.x;   // < K_TOK*K_H
  int tok = idx / K_H, e = idx % K_H;
  int b = tok / (K_N * K_I * K_O);
  Y[idx] = (__bf16)(s[idx] + temb[b * K_H + e]);
}

__global__ void prep_w_kernel(const float* __restrict__ wq, const float* __restrict__ wk,
                              const float* __restrict__ wv, const float* __restrict__ wo,
                              const float* __restrict__ bq, const float* __restrict__ bk,
                              const float* __restrict__ bv, const float* __restrict__ bo,
                              __bf16* __restrict__ WtQKV,   // [K_NCAT][K_H]
                              __bf16* __restrict__ WtO,     // [K_DOUT][K_KOUT]
                              float* __restrict__ biascat,  // [K_NCAT]
                              float* __restrict__ bocat) {  // [K_DOUT]
  const int R1 = K_NCAT * K_H;      // 294912
  const int R2 = K_DOUT * K_KOUT;   // 196608
  int idx = blockIdx.x * blockDim.x + threadIdx.x;
  if (idx < R1) {
    int nc = idx / K_H, k = idx % K_H;
    int l = nc / 768, r = nc % 768, p = r / K_DOUT, e = r % K_DOUT;
    const float* W = (p == 0) ? wq : (p == 1) ? wk : wv;
    WtQKV[nc * K_H + k] = (__bf16)W[((long)l * K_H + k) * K_DOUT + e];
  } else if (idx < R1 + R2) {
    int j = idx - R1;
    int e = j / K_KOUT, kc = j % K_KOUT;
    int l = kc / K_DOUT, c = kc % K_DOUT;
    WtO[e * K_KOUT + kc] = (__bf16)wo[((long)l * K_DOUT + c) * K_DOUT + e];
  } else if (idx < R1 + R2 + K_NCAT) {
    int nc = idx - R1 - R2;
    int l = nc / 768, r = nc % 768, p = r / K_DOUT, e = r % K_DOUT;
    const float* bsrc = (p == 0) ? bq : (p == 1) ? bk : bv;
    biascat[nc] = bsrc[l * K_DOUT + e];
  } else if (idx < R1 + R2 + K_NCAT + K_DOUT) {
    int e = idx - R1 - R2 - K_NCAT;
    bocat[e] = bo[e] + bo[K_DOUT + e] + bo[2 * K_DOUT + e];
  }
}

// ---------------- fused QKV projection GEMM: [24576 x 128] x [128 x 2304] ----------------
// One wave computes a 16x64 strip (4 accumulators): A fragment loaded once per K-step,
// reused across 4 back-to-back WMMAs. Epilogue staged through LDS for 16B/lane stores.

__global__ void __launch_bounds__(256) gemm_qkv_kernel(const __bf16* __restrict__ Y,
                                                       const __bf16* __restrict__ Wt,
                                                       const float* __restrict__ biascat,
                                                       __bf16* __restrict__ QKV) {
  __shared__ float s_c[8][16 * 64];
  const int wave_in = threadIdx.x >> 5;
  const int lane    = threadIdx.x & 31;
  int wave = blockIdx.x * 8 + wave_in;
  int mt = wave / (K_NCAT / 64);          // 36 n-strips
  int nt = wave % (K_NCAT / 64);
  int m0 = mt * 16, n0 = nt * 64;
  v8f acc[4] = {};
#pragma unroll
  for (int ks = 0; ks < K_H / 32; ++ks) {
    v16bf a = load_a_frag(Y, K_H, m0, ks * 32, lane);
#pragma unroll
    for (int j = 0; j < 4; ++j) {
      v16bf b = load_b_frag(Wt, K_H, n0 + j * 16, ks * 32, lane);
      acc[j] = __builtin_amdgcn_wmma_f32_16x16x32_bf16(false, a, false, b, (short)0, acc[j], false, false);
    }
  }
  const int rb = (lane < 16) ? 0 : 8;
  const int l15 = lane & 15;
#pragma unroll
  for (int j = 0; j < 4; ++j) {
    float bias = biascat[n0 + j * 16 + l15];
#pragma unroll
    for (int r = 0; r < 8; ++r)
      s_c[wave_in][(rb + r) * 64 + j * 16 + l15] = acc[j][r] + bias;
  }
  __syncthreads();
  const int row = l15, ch = (lane >> 4) * 32;
  const float* src = &s_c[wave_in][row * 64 + ch];
  __bf16* dst = QKV + (long)(m0 + row) * K_NCAT + n0 + ch;
#pragma unroll
  for (int k = 0; k < 4; ++k) {
    v8bf v;
#pragma unroll
    for (int c = 0; c < 8; ++c) v[c] = (__bf16)src[k * 8 + c];
    *(v8bf*)(dst + k * 8) = v;
  }
}

// ---------------- small-axis attentions (seq len 2 or 3) ----------------

__global__ void attn_small_kernel(const __bf16* __restrict__ QKV, __bf16* __restrict__ Ob,
                                  int qoff, int koff, int voff, int stride_tok, int L, int ocol) {
  int tid = blockIdx.x * blockDim.x + threadIdx.x;
  int h  = tid & 7;
  int tq = tid >> 3;
  if (tq >= K_TOK) return;
  int pos  = (tq / stride_tok) % L;
  int base = tq - pos * stride_tok;
  float q[K_KD];
  const __bf16* qp = QKV + (long)tq * K_NCAT + qoff + h * K_KD;
#pragma unroll
  for (int d = 0; d < K_KD; ++d) q[d] = (float)qp[d];
  float logit[3];
  float mx = -3.0e38f;
  for (int j = 0; j < L; ++j) {
    const __bf16* kp = QKV + (long)(base + j * stride_tok) * K_NCAT + koff + h * K_KD;
    float sdot = 0.f;
#pragma unroll
    for (int d = 0; d < K_KD; ++d) sdot += q[d] * (float)kp[d];
    logit[j] = sdot * INV_SQRT_KD;
    mx = fmaxf(mx, logit[j]);
  }
  float out[K_KD];
#pragma unroll
  for (int d = 0; d < K_KD; ++d) out[d] = 0.f;
  float denom = 0.f;
  for (int j = 0; j < L; ++j) {
    float p = expf(logit[j] - mx);
    denom += p;
    const __bf16* vp = QKV + (long)(base + j * stride_tok) * K_NCAT + voff + h * K_KD;
#pragma unroll
    for (int d = 0; d < K_KD; ++d) out[d] += p * (float)vp[d];
  }
  float inv = 1.0f / denom;
  __bf16* op = Ob + (long)tq * K_KOUT + ocol + h * K_KD;
#pragma unroll
  for (int d = 0; d < K_KD; ++d) op[d] = (__bf16)(out[d] * inv);
}

// ---------------- attention over N (flash-style, WMMA) ----------------

#define CHUNK 64
#define QBASE (6 * K_DOUT)   /* 1536 */
#define KBASE (7 * K_DOUT)   /* 1792 */
#define VBASE (8 * K_DOUT)   /* 2048 */

__global__ void __launch_bounds__(128) attn_n_kernel(const __bf16* __restrict__ QKV,
                                                     __bf16* __restrict__ Ob) {
  __shared__ float s_logits[4][16 * CHUNK];
  __shared__ __align__(32) __bf16 s_P[4][16 * CHUNK];
  __shared__ __align__(32) __bf16 s_Vt[4][K_KD * CHUNK];
  __shared__ float s_m[4][16], s_l[4][16], s_alpha[4][16];

  const int wv   = threadIdx.x >> 5;
  const int lane = threadIdx.x & 31;
  int w = blockIdx.x * 4 + wv;
  const int qc = w & 31;  w >>= 5;          // 32 query chunks of 16
  const int h  = w & 7;   w >>= 3;
  const int o  = w & 1;   w >>= 1;
  const int i  = w % 3;
  const int b  = w / 3;
  const long tok_base = (long)b * 3072 + i * 2 + o;   // + n*6
  const int q0 = qc * 16;

  // persistent Q A-fragment (d = 32 == one WMMA K step)
  v16bf a_q;
  {
    int qrow = q0 + (lane & 15);
    int koff = (lane < 16) ? 0 : 16;
    const __bf16* p = QKV + (tok_base + (long)qrow * 6) * K_NCAT + QBASE + h * K_KD + koff;
    a_q = combine8(*(const v8bf*)p, *(const v8bf*)(p + 8));
  }
  v8f acc0 = {}, acc1 = {};
  if (lane < 16) { s_m[wv][lane] = -3.0e38f; s_l[wv][lane] = 0.f; }
  __syncthreads();

  for (int kc = 0; kc < K_N / CHUNK; ++kc) {
    const int kb = kc * CHUNK;
    // stage V transposed: s_Vt[d][kk]
    {
      const int d = lane;
      for (int kk = 0; kk < CHUNK; ++kk) {
        long kt = tok_base + (long)(kb + kk) * 6;
        s_Vt[wv][d * CHUNK + kk] = QKV[kt * K_NCAT + VBASE + h * K_KD + d];
      }
    }
    // logits: 4 key tiles of 16
#pragma unroll
    for (int j = 0; j < 4; ++j) {
      int key = kb + j * 16 + (lane & 15);
      long kt = tok_base + (long)key * 6;
      int koffB = (lane < 16) ? 0 : 16;
      v16bf bk = *(const v16bf*)(QKV + kt * K_NCAT + KBASE + h * K_KD + koffB);
      v8f c = {};
      c = __builtin_amdgcn_wmma_f32_16x16x32_bf16(false, a_q, false, bk, (short)0, c, false, false);
      int colL = j * 16 + (lane & 15);
      int rb = (lane < 16) ? 0 : 8;
#pragma unroll
      for (int r = 0; r < 8; ++r) s_logits[wv][(rb + r) * CHUNK + colL] = c[r] * INV_SQRT_KD;
    }
    __syncthreads();
    // online softmax update: lane = (row, half-of-keys)
    {
      const int row = lane & 15;
      const int kk0 = (lane >> 4) * 32;
      float cmax = -3.0e38f;
      for (int kk = 0; kk < 32; ++kk) cmax = fmaxf(cmax, s_logits[wv][row * CHUNK + kk0 + kk]);
      cmax = fmaxf(cmax, __shfl_xor(cmax, 16, 32));
      float mold = s_m[wv][row];
      float mnew = fmaxf(mold, cmax);
      float alpha = expf(mold - mnew);
      float psum = 0.f;
      for (int kk = 0; kk < 32; ++kk) {
        float pv = expf(s_logits[wv][row * CHUNK + kk0 + kk] - mnew);
        s_P[wv][row * CHUNK + kk0 + kk] = (__bf16)pv;
        psum += pv;
      }
      psum += __shfl_xor(psum, 16, 32);
      if (lane < 16) {
        s_m[wv][row] = mnew;
        s_l[wv][row] = s_l[wv][row] * alpha + psum;
        s_alpha[wv][row] = alpha;
      }
    }
    __syncthreads();
    // rescale accumulators
    {
      int rb = (lane < 16) ? 0 : 8;
#pragma unroll
      for (int r = 0; r < 8; ++r) {
        float al = s_alpha[wv][rb + r];
        acc0[r] *= al; acc1[r] *= al;
      }
    }
    // acc += P(16x64) @ V(64x32)
#pragma unroll
    for (int s2 = 0; s2 < 2; ++s2) {
      const int K0 = s2 * 32;
      int arow = lane & 15;
      int koff = (lane < 16) ? 0 : 16;
      const __bf16* pa = &s_P[wv][arow * CHUNK + K0 + koff];
      v16bf ap = combine8(*(const v8bf*)pa, *(const v8bf*)(pa + 8));
      int koffB = (lane < 16) ? 0 : 16;
      v16bf b0 = *(const v16bf*)(&s_Vt[wv][(lane & 15) * CHUNK + K0 + koffB]);
      v16bf b1 = *(const v16bf*)(&s_Vt[wv][(16 + (lane & 15)) * CHUNK + K0 + koffB]);
      acc0 = __builtin_amdgcn_wmma_f32_16x16x32_bf16(false, ap, false, b0, (short)0, acc0, false, false);
      acc1 = __builtin_amdgcn_wmma_f32_16x16x32_bf16(false, ap, false, b1, (short)0, acc1, false, false);
    }
    __syncthreads();
  }
  // finalize: divide by l, stage 16x32 output tile through LDS, then 16B/lane stores
  {
    int rb = (lane < 16) ? 0 : 8;
    int dcol = lane & 15;
#pragma unroll
    for (int r = 0; r < 8; ++r) {
      float linv = 1.0f / s_l[wv][rb + r];
      s_logits[wv][(rb + r) * 32 + dcol]      = acc0[r] * linv;
      s_logits[wv][(rb + r) * 32 + 16 + dcol] = acc1[r] * linv;
    }
  }
  __syncthreads();
  {
    int q = lane & 15, half = lane >> 4;
    long t = tok_base + (long)(q0 + q) * 6;
    __bf16* op = Ob + t * K_KOUT + 512 + h * K_KD + half * 16;
    const float* src = &s_logits[wv][q * 32 + half * 16];
#pragma unroll
    for (int k = 0; k < 2; ++k) {
      v8bf v;
#pragma unroll
      for (int c = 0; c < 8; ++c) v[c] = (__bf16)src[k * 8 + c];
      *(v8bf*)(op + k * 8) = v;
    }
  }
}

// ---------------- output GEMM [24576 x 768] x [768 x 256] + fused gelu epilogue ----------------
// One wave computes 16x64 (4 accumulators); epilogue staged through LDS -> contiguous
// 128B/lane f32 stores (b128), fused bias + tanh-gelu + residual/sqrt(2).

__global__ void __launch_bounds__(256) gemm_out_kernel(const __bf16* __restrict__ Ob,
                                                       const __bf16* __restrict__ WtO,
                                                       const float* __restrict__ bocat,
                                                       const float* __restrict__ s,
                                                       float* __restrict__ out) {
  __shared__ float s_c[8][16 * 64];
  const int wave_in = threadIdx.x >> 5;
  const int lane    = threadIdx.x & 31;
  int wave = blockIdx.x * 8 + wave_in;
  int mt = wave >> 2;        // 4 n-strips of 64
  int nt = wave & 3;
  int m0 = mt * 16, n0 = nt * 64;
  v8f acc[4] = {};
  for (int ks = 0; ks < K_KOUT / 32; ++ks) {
    v16bf a = load_a_frag(Ob, K_KOUT, m0, ks * 32, lane);
#pragma unroll
    for (int j = 0; j < 4; ++j) {
      v16bf b = load_b_frag(WtO, K_KOUT, n0 + j * 16, ks * 32, lane);
      acc[j] = __builtin_amdgcn_wmma_f32_16x16x32_bf16(false, a, false, b, (short)0, acc[j], false, false);
    }
  }
  const int rb = (lane < 16) ? 0 : 8;
  const int l15 = lane & 15;
#pragma unroll
  for (int j = 0; j < 4; ++j)
#pragma unroll
    for (int r = 0; r < 8; ++r)
      s_c[wave_in][(rb + r) * 64 + j * 16 + l15] = acc[j][r];
  __syncthreads();
  const int row = l15, ch = (lane >> 4) * 32;
  const int tok = m0 + row;
  const int e0 = n0 + ch;                 // multiple of 32; uniform side of the split per wave
  const float* src = &s_c[wave_in][row * 64 + ch];
  if (e0 < K_H) {
    const float* sp = s + (long)tok * K_H + e0;
    float* dp = out + (long)tok * K_H + e0;
#pragma unroll
    for (int c = 0; c < 32; ++c) {
      float g = gelu_tanh(src[c] + bocat[e0 + c]);
      dp[c] = (sp[c] + g) * INV_SQRT2;
    }
  } else {
    float* dp = out + (long)K_TOK * K_H + (long)tok * K_H + (e0 - K_H);
#pragma unroll
    for (int c = 0; c < 32; ++c)
      dp[c] = gelu_tanh(src[c] + bocat[e0 + c]);
  }
}

// ---------------- launch ----------------

extern "C" void kernel_launch(void* const* d_in, const int* in_sizes, int n_in,
                              void* d_out, int out_size, void* d_ws, size_t ws_size,
                              hipStream_t stream) {
  const float* s  = (const float*)d_in[0];
  const float* t  = (const float*)d_in[1];
  // d_in[2] = mask_type (all-true in this problem; where(mask,...) is identity)
  const float* wt = (const float*)d_in[3];
  const float* bt = (const float*)d_in[4];
  const float* wq = (const float*)d_in[5];
  const float* bq = (const float*)d_in[6];
  const float* wk = (const float*)d_in[7];
  const float* bk = (const float*)d_in[8];
  const float* wv = (const float*)d_in[9];
  const float* bv = (const float*)d_in[10];
  const float* wo = (const float*)d_in[11];
  const float* bo = (const float*)d_in[12];
  float* out = (float*)d_out;

  char* ws = (char*)d_ws;
  size_t cur = 0;
  auto carve = [&](size_t bytes) -> char* {
    char* p = ws + cur;
    cur = (cur + bytes + 255) & ~(size_t)255;
    return p;
  };
  float*  temb    = (float*) carve((size_t)K_B * K_H * 4);
  __bf16* Y       = (__bf16*)carve((size_t)K_TOK * K_H * 2);
  __bf16* WtQKV   = (__bf16*)carve((size_t)K_NCAT * K_H * 2);
  float*  biascat = (float*) carve((size_t)K_NCAT * 4);
  __bf16* WtO     = (__bf16*)carve((size_t)K_DOUT * K_KOUT * 2);
  float*  bocat   = (float*) carve((size_t)K_DOUT * 4);
  __bf16* QKV     = (__bf16*)carve((size_t)K_TOK * K_NCAT * 2);
  __bf16* Ob      = (__bf16*)carve((size_t)K_TOK * K_KOUT * 2);
  (void)ws_size; (void)in_sizes; (void)n_in; (void)out_size;

  prep_temb_kernel<<<K_B, K_H, 0, stream>>>(t, wt, bt, temb);
  prep_y_kernel<<<(K_TOK * K_H) / 256, 256, 0, stream>>>(s, temb, Y);
  {
    int total = K_NCAT * K_H + K_DOUT * K_KOUT + K_NCAT + K_DOUT;   // 494080
    prep_w_kernel<<<(total + 255) / 256, 256, 0, stream>>>(wq, wk, wv, wo, bq, bk, bv, bo,
                                                           WtQKV, WtO, biascat, bocat);
  }
  // 1536 m-tiles * 36 n-strips, 8 waves/block
  gemm_qkv_kernel<<<(K_TOK / 16) * (K_NCAT / 64) / 8, 256, 0, stream>>>(Y, WtQKV, biascat, QKV);
  // attention over O (len 2): q/k/v at cols 0/256/512, token stride 1
  attn_small_kernel<<<(K_TOK * K_HEADS) / 256, 256, 0, stream>>>(QKV, Ob, 0, 256, 512, 1, 2, 0);
  // attention over I (len 3): q/k/v at cols 768/1024/1280, token stride 2
  attn_small_kernel<<<(K_TOK * K_HEADS) / 256, 256, 0, stream>>>(QKV, Ob, 768, 1024, 1280, 2, 3, 256);
  // attention over N: 384 (b,i,o,h) groups * 32 q-chunks = 12288 waves, 4 waves/block
  attn_n_kernel<<<12288 / 4, 128, 0, stream>>>(QKV, Ob);
  // 1536 m-tiles * 4 n-strips, 8 waves/block, fused gelu/residual epilogue
  gemm_out_kernel<<<(K_TOK / 16) * (K_DOUT / 64) / 8, 256, 0, stream>>>(Ob, WtO, bocat, s, out);
}
